// MambaBackboneBlock_4398046511440
// MI455X (gfx1250) — matrix operand order repaired and checked by reading, not compiled, not run on hardware
//
#include <hip/hip_runtime.h>

typedef __attribute__((ext_vector_type(16))) __bf16 v16bf;
typedef __attribute__((ext_vector_type(8)))  float  v8f;
typedef __attribute__((ext_vector_type(8)))  __bf16 bf16x8;
typedef int v4i_ __attribute__((vector_size(16)));   // matches builtin's V4i param type

#define BB 4
#define LL 4096
#define DD 1024
#define HH 2048
#define BL (BB*LL)

#ifndef __has_builtin
#define __has_builtin(x) 0
#endif
#if __has_builtin(__builtin_amdgcn_global_load_async_to_lds_b128) && \
    __has_builtin(__builtin_amdgcn_s_wait_asynccnt)
#define USE_ASYNC 1
#else
#define USE_ASYNC 0
#endif

#define GLOBAL_AS __attribute__((address_space(1)))
#define LDS_AS    __attribute__((address_space(3)))

// ---------------------------------------------------------------- utilities
__global__ __launch_bounds__(256) void cvt_kernel(const float* __restrict__ src,
                                                  __bf16* __restrict__ dst,
                                                  int rows, int cols, int dstStride, int dstOff) {
    long long i = (long long)blockIdx.x * 256 + threadIdx.x;
    long long n = (long long)rows * cols;
    if (i >= n) return;
    int r = (int)(i / cols), c = (int)(i % cols);
    dst[(size_t)r * dstStride + dstOff + c] = (__bf16)src[i];
}

__global__ __launch_bounds__(256) void concat3_kernel(const float* __restrict__ a,
                                                      const float* __restrict__ b,
                                                      const float* __restrict__ c,
                                                      float* __restrict__ dst) {
    int i = blockIdx.x * 256 + threadIdx.x;
    if (i >= 3 * DD) return;
    dst[i] = (i < DD) ? a[i] : (i < 2 * DD) ? b[i - DD] : c[i - 2 * DD];
}

// ---------------------------------------------------------------- layernorm
__global__ __launch_bounds__(256) void ln_kernel(const float* __restrict__ x,
                                                 const float* __restrict__ g,
                                                 const float* __restrict__ b,
                                                 __bf16* __restrict__ out) {
    __shared__ float red[256];
    const int row = blockIdx.x;
    const int tid = threadIdx.x;
    const float* xr = x + (size_t)row * DD;
    float vals[4];
    float s = 0.f;
#pragma unroll
    for (int i = 0; i < 4; ++i) { vals[i] = xr[tid + i * 256]; s += vals[i]; }
    red[tid] = s; __syncthreads();
    for (int off = 128; off > 0; off >>= 1) {
        if (tid < off) red[tid] += red[tid + off];
        __syncthreads();
    }
    const float mean = red[0] * (1.f / DD);
    __syncthreads();
    float vs = 0.f;
#pragma unroll
    for (int i = 0; i < 4; ++i) { float d = vals[i] - mean; vs += d * d; }
    red[tid] = vs; __syncthreads();
    for (int off = 128; off > 0; off >>= 1) {
        if (tid < off) red[tid] += red[tid + off];
        __syncthreads();
    }
    const float rstd = rsqrtf(red[0] * (1.f / DD) + 1e-5f);
#pragma unroll
    for (int i = 0; i < 4; ++i) {
        int c = tid + i * 256;
        out[(size_t)row * DD + c] = (__bf16)((vals[i] - mean) * rstd * g[c] + b[c]);
    }
}

// -------------------------------------------------- depthwise conv (K=3, SAME)
__global__ __launch_bounds__(256) void conv_kernel(const float* __restrict__ proj,
                                                   const float* __restrict__ dwW,
                                                   const float* __restrict__ dwb,
                                                   float* __restrict__ mixedF,
                                                   __bf16* __restrict__ mixedH) {
    long long idx = (long long)blockIdx.x * 256 + threadIdx.x;
    if (idx >= (long long)BL * DD) return;
    int d = (int)(idx % DD);
    long long row = idx / DD;          // token index b*L + t
    int t = (int)(row % LL);
    float acc = dwb[d];
    const float w0 = dwW[d * 3 + 0], w1 = dwW[d * 3 + 1], w2 = dwW[d * 3 + 2];
    if (t > 0)      acc += proj[(row - 1) * (2 * DD) + d] * w0;
    acc += proj[row * (2 * DD) + d] * w1;
    if (t < LL - 1) acc += proj[(row + 1) * (2 * DD) + d] * w2;
    mixedF[idx] = acc;
    mixedH[idx] = (__bf16)acc;
}

// -------------------------------------------- chunked linear-recurrence scan
// s_t = exp(dt*a)*s_{t-1} + dt*b*m ; y = c*s + d_skip*m ; out = y*sigmoid(gate)
#define CH 64          // chunks along L
#define CT (LL / CH)   // steps per chunk

__global__ __launch_bounds__(256) void scan_pass1(const float* __restrict__ dtbc,
                                                  const float* __restrict__ mixed,
                                                  const float* __restrict__ a_log,
                                                  float* __restrict__ Pbuf,
                                                  float* __restrict__ rbuf) {
    int id = blockIdx.x * 256 + threadIdx.x;       // (b*CH + c)*DD + d
    if (id >= BB * CH * DD) return;
    const int d = id % DD;
    const int c = (id / DD) % CH;
    const int b = id / (DD * CH);
    const float a = -expf(a_log[d]);
    float P = 1.f, r = 0.f;
    for (int t = c * CT; t < (c + 1) * CT; ++t) {
        size_t row = (size_t)b * LL + t;
        float dtv = dtbc[row * (3 * DD) + d];
        float btv = dtbc[row * (3 * DD) + DD + d];
        float m   = mixed[row * DD + d];
        float decay = expf(dtv * a);
        r = decay * r + dtv * btv * m;
        P *= decay;
    }
    Pbuf[id] = P;
    rbuf[id] = r;
}

__global__ __launch_bounds__(256) void scan_pass2(const float* __restrict__ Pbuf,
                                                  const float* __restrict__ rbuf,
                                                  float* __restrict__ Sbuf) {
    int id = blockIdx.x * 256 + threadIdx.x;       // b*DD + d
    if (id >= BB * DD) return;
    const int d = id % DD;
    const int b = id / DD;
    float s = 0.f;
    for (int c = 0; c < CH; ++c) {
        size_t i = ((size_t)b * CH + c) * DD + d;
        Sbuf[i] = s;
        s = Pbuf[i] * s + rbuf[i];
    }
}

__global__ __launch_bounds__(256) void scan_pass3(const float* __restrict__ dtbc,
                                                  const float* __restrict__ mixed,
                                                  const float* __restrict__ proj,
                                                  const float* __restrict__ a_log,
                                                  const float* __restrict__ d_skip,
                                                  const float* __restrict__ Sbuf,
                                                  __bf16* __restrict__ yH) {
    int id = blockIdx.x * 256 + threadIdx.x;       // (b*CH + c)*DD + d
    if (id >= BB * CH * DD) return;
    const int d = id % DD;
    const int c = (id / DD) % CH;
    const int b = id / (DD * CH);
    const float a = -expf(a_log[d]);
    const float dsk = d_skip[d];
    float s = Sbuf[id];
    for (int t = c * CT; t < (c + 1) * CT; ++t) {
        size_t row = (size_t)b * LL + t;
        float dtv = dtbc[row * (3 * DD) + d];
        float btv = dtbc[row * (3 * DD) + DD + d];
        float ctv = dtbc[row * (3 * DD) + 2 * DD + d];
        float m   = mixed[row * DD + d];
        float gt  = proj[row * (2 * DD) + DD + d];
        float decay = expf(dtv * a);
        s = decay * s + dtv * btv * m;
        float y = ctv * s + dsk * m;
        yH[row * DD + d] = (__bf16)(y * (1.f / (1.f + expf(-gt))));
    }
}

// ---------------------------------------------------------------- WMMA GEMM
// C[M,N] = A[M,K](bf16) @ W[K,N](bf16) + bias, epilogues:
//   EPI 0: f32 out ; EPI 1: softplus/tanh split ; EPI 2: +resid ; EPI 3: GELU->bf16
#define BM 128
#define BN 128
#define BK 32

template <int EPI>
__global__ __launch_bounds__(256) void gemm_bf16_kernel(const __bf16* __restrict__ A,
                                                        const __bf16* __restrict__ W,
                                                        const float* __restrict__ bias,
                                                        const float* __restrict__ resid,
                                                        float* __restrict__ outF,
                                                        __bf16* __restrict__ outH,
                                                        int M, int N, int K) {
    __shared__ __bf16 As[2][BM][BK + 8];
    __shared__ __bf16 Bs[2][BK][BN + 8];

    const int tid   = threadIdx.x;
    const int lane  = tid & 31;
    const int wave  = tid >> 5;      // 0..7
    const int waveM = wave >> 1;     // 0..3 -> 32-row slice
    const int waveN = wave & 1;      // 0..1 -> 64-col slice
    const int blockM = blockIdx.y * BM;
    const int blockN = blockIdx.x * BN;
    const int r  = lane & 15;
    const int hi = lane >> 4;

    v8f acc[2][4];
#pragma unroll
    for (int mt = 0; mt < 2; ++mt)
#pragma unroll
        for (int nt = 0; nt < 4; ++nt) acc[mt][nt] = (v8f){};

    auto load_tiles = [&](int buf, int kt) {
#pragma unroll
        for (int i = 0; i < 2; ++i) {           // A: 128x32 = 512 x 16B units
            int u = tid + i * 256;
            int row = u >> 2, col = (u & 3) << 3;
            const __bf16* gsrc = A + (size_t)(blockM + row) * K + kt * BK + col;
            __bf16* ldst = &As[buf][row][col];
#if USE_ASYNC
            __builtin_amdgcn_global_load_async_to_lds_b128(
                (GLOBAL_AS v4i_*)(uintptr_t)gsrc,
                (LDS_AS v4i_*)(unsigned)(uintptr_t)ldst, 0, 0);
#else
            *(bf16x8*)ldst = *(const bf16x8*)gsrc;
#endif
        }
#pragma unroll
        for (int i = 0; i < 2; ++i) {           // B: 32x128 = 512 x 16B units
            int u = tid + i * 256;
            int row = u >> 4, col = (u & 15) << 3;
            const __bf16* gsrc = W + (size_t)(kt * BK + row) * N + blockN + col;
            __bf16* ldst = &Bs[buf][row][col];
#if USE_ASYNC
            __builtin_amdgcn_global_load_async_to_lds_b128(
                (GLOBAL_AS v4i_*)(uintptr_t)gsrc,
                (LDS_AS v4i_*)(unsigned)(uintptr_t)ldst, 0, 0);
#else
            *(bf16x8*)ldst = *(const bf16x8*)gsrc;
#endif
        }
    };

    const int nK = K / BK;
    load_tiles(0, 0);
    for (int kt = 0; kt < nK; ++kt) {
        const int cur = kt & 1;
#if USE_ASYNC
        __builtin_amdgcn_s_wait_asynccnt(0);
#endif
        __syncthreads();
        if (kt + 1 < nK) load_tiles(cur ^ 1, kt + 1);

        // fragments per ISA layout: A lane r holds k = 8*hi + i + (i&8);
        // B lane r holds column n = r, same k order as A
        v16bf afrag[2];
#pragma unroll
        for (int mt = 0; mt < 2; ++mt) {
            const __bf16* Ar = &As[cur][waveM * 32 + mt * 16 + r][hi * 8];
#pragma unroll
            for (int i = 0; i < 16; ++i) afrag[mt][i] = Ar[i + (i & 8)];
        }
#pragma unroll
        for (int nt = 0; nt < 4; ++nt) {
            v16bf bfrag;
            const __bf16* Br = &Bs[cur][hi * 8][waveN * 64 + nt * 16 + r];
#pragma unroll
            for (int i = 0; i < 16; ++i) bfrag[i] = Br[(i + (i & 8)) * (BN + 8)];
#pragma unroll
            for (int mt = 0; mt < 2; ++mt)
                acc[mt][nt] = __builtin_amdgcn_wmma_f32_16x16x32_bf16(
                    false, afrag[mt], false, bfrag, (short)0, acc[mt][nt], false, false);
        }
    }

    // ---- epilogue
#pragma unroll
    for (int mt = 0; mt < 2; ++mt) {
#pragma unroll
        for (int nt = 0; nt < 4; ++nt) {
            const int n = blockN + waveN * 64 + nt * 16 + r;
            const float bia = bias[n];
#pragma unroll
            for (int v = 0; v < 8; ++v) {
                const int m = blockM + waveM * 32 + mt * 16 + hi * 8 + v;
                const size_t idx = (size_t)m * N + n;
                float val = acc[mt][nt][v] + bia;
                if (EPI == 0) {
                    outF[idx] = val;
                } else if (EPI == 1) {
                    if (n < N / 3) outF[idx] = (val > 20.f) ? val : log1pf(expf(val));
                    else           outF[idx] = tanhf(val);
                } else if (EPI == 2) {
                    outF[idx] = val + resid[idx];
                } else {
                    outH[idx] = (__bf16)(0.5f * val * (1.f + erff(val * 0.70710678118f)));
                }
            }
        }
    }
}

// ---------------------------------------------------------------- launcher
extern "C" void kernel_launch(void* const* d_in, const int* in_sizes, int n_in,
                              void* d_out, int out_size, void* d_ws, size_t ws_size,
                              hipStream_t stream) {
    (void)in_sizes; (void)n_in; (void)out_size; (void)ws_size;
    const float* x      = (const float*)d_in[0];
    const float* ln1_g  = (const float*)d_in[1];
    const float* ln1_b  = (const float*)d_in[2];
    const float* in_W   = (const float*)d_in[3];
    const float* in_b   = (const float*)d_in[4];
    const float* dw_W   = (const float*)d_in[5];
    const float* dw_b   = (const float*)d_in[6];
    const float* dt_W   = (const float*)d_in[7];
    const float* dt_b   = (const float*)d_in[8];
    const float* b_W    = (const float*)d_in[9];
    const float* b_b    = (const float*)d_in[10];
    const float* c_W    = (const float*)d_in[11];
    const float* c_b    = (const float*)d_in[12];
    const float* a_log  = (const float*)d_in[13];
    const float* d_skip = (const float*)d_in[14];
    const float* out_W  = (const float*)d_in[15];
    const float* out_b  = (const float*)d_in[16];
    const float* ln2_g  = (const float*)d_in[17];
    const float* ln2_b  = (const float*)d_in[18];
    const float* f1_W   = (const float*)d_in[19];
    const float* f1_b   = (const float*)d_in[20];
    const float* f2_W   = (const float*)d_in[21];
    const float* f2_b   = (const float*)d_in[22];
    float* outp = (float*)d_out;

    char* ws = (char*)d_ws;
    size_t off = 0;
    auto alloc = [&](size_t bytes) -> void* {
        void* p = ws + off;
        off += (bytes + 255) & ~(size_t)255;
        return p;
    };
    __bf16* nxH     = (__bf16*)alloc((size_t)BL * DD * 2);
    __bf16* inWb    = (__bf16*)alloc((size_t)DD * 2 * DD * 2);
    __bf16* dtbcWb  = (__bf16*)alloc((size_t)DD * 3 * DD * 2);
    __bf16* outWb   = (__bf16*)alloc((size_t)DD * DD * 2);
    __bf16* f1Wb    = (__bf16*)alloc((size_t)DD * HH * 2);
    __bf16* f2Wb    = (__bf16*)alloc((size_t)HH * DD * 2);
    float*  dtbcB   = (float*)alloc((size_t)3 * DD * 4);
    float*  projF   = (float*)alloc((size_t)BL * 2 * DD * 4);
    float*  mixedF  = (float*)alloc((size_t)BL * DD * 4);
    __bf16* mixedH  = (__bf16*)alloc((size_t)BL * DD * 2);
    float*  dtbcF   = (float*)alloc((size_t)BL * 3 * DD * 4);
    __bf16* yH      = (__bf16*)alloc((size_t)BL * DD * 2);
    float*  x2F     = (float*)alloc((size_t)BL * DD * 4);
    __bf16* hH      = (__bf16*)alloc((size_t)BL * DD * 2);
    __bf16* gH      = (__bf16*)alloc((size_t)BL * HH * 2);
    float*  Pbuf    = (float*)alloc((size_t)BB * CH * DD * 4);
    float*  rbuf    = (float*)alloc((size_t)BB * CH * DD * 4);
    float*  Sbuf    = (float*)alloc((size_t)BB * CH * DD * 4);

    auto nb = [](long long n) { return (unsigned)((n + 255) / 256); };

    // weight conversions (f32 -> bf16), dt/b/c fused into [D, 3D]
    cvt_kernel<<<nb((long long)DD * 2 * DD), 256, 0, stream>>>(in_W, inWb, DD, 2 * DD, 2 * DD, 0);
    cvt_kernel<<<nb((long long)DD * DD), 256, 0, stream>>>(dt_W, dtbcWb, DD, DD, 3 * DD, 0);
    cvt_kernel<<<nb((long long)DD * DD), 256, 0, stream>>>(b_W,  dtbcWb, DD, DD, 3 * DD, DD);
    cvt_kernel<<<nb((long long)DD * DD), 256, 0, stream>>>(c_W,  dtbcWb, DD, DD, 3 * DD, 2 * DD);
    cvt_kernel<<<nb((long long)DD * DD), 256, 0, stream>>>(out_W, outWb, DD, DD, DD, 0);
    cvt_kernel<<<nb((long long)DD * HH), 256, 0, stream>>>(f1_W, f1Wb, DD, HH, HH, 0);
    cvt_kernel<<<nb((long long)HH * DD), 256, 0, stream>>>(f2_W, f2Wb, HH, DD, DD, 0);
    concat3_kernel<<<nb(3 * DD), 256, 0, stream>>>(dt_b, b_b, c_b, dtbcB);

    // LN1 -> nx (bf16)
    ln_kernel<<<BL, 256, 0, stream>>>(x, ln1_g, ln1_b, nxH);

    // proj = nx @ in_W + in_b                 [BL, 2D]
    gemm_bf16_kernel<0><<<dim3(2 * DD / BN, BL / BM), 256, 0, stream>>>(
        nxH, inWb, in_b, nullptr, projF, nullptr, BL, 2 * DD, DD);

    // depthwise conv over L (SAME)
    conv_kernel<<<nb((long long)BL * DD), 256, 0, stream>>>(projF, dw_W, dw_b, mixedF, mixedH);

    // [dt|b|c] = act(mixed @ [dt_W|b_W|c_W] + bias)   [BL, 3D]
    gemm_bf16_kernel<1><<<dim3(3 * DD / BN, BL / BM), 256, 0, stream>>>(
        mixedH, dtbcWb, dtbcB, nullptr, dtbcF, nullptr, BL, 3 * DD, DD);

    // chunk-parallel selective scan + gate
    scan_pass1<<<nb((long long)BB * CH * DD), 256, 0, stream>>>(dtbcF, mixedF, a_log, Pbuf, rbuf);
    scan_pass2<<<nb(BB * DD), 256, 0, stream>>>(Pbuf, rbuf, Sbuf);
    scan_pass3<<<nb((long long)BB * CH * DD), 256, 0, stream>>>(dtbcF, mixedF, projF, a_log,
                                                                d_skip, Sbuf, yH);

    // x2 = x + y @ out_W + out_b
    gemm_bf16_kernel<2><<<dim3(DD / BN, BL / BM), 256, 0, stream>>>(
        yH, outWb, out_b, x, x2F, nullptr, BL, DD, DD);

    // LN2 -> h (bf16)
    ln_kernel<<<BL, 256, 0, stream>>>(x2F, ln2_g, ln2_b, hH);

    // g = gelu(h @ f1_W + f1_b)  (bf16 out)
    gemm_bf16_kernel<3><<<dim3(HH / BN, BL / BM), 256, 0, stream>>>(
        hH, f1Wb, f1_b, nullptr, nullptr, gH, BL, HH, DD);

    // out = x2 + g @ f2_W + f2_b
    gemm_bf16_kernel<2><<<dim3(DD / BN, BL / BM), 256, 0, stream>>>(
        gH, f2Wb, f2_b, x2F, outp, nullptr, BL, DD, HH);
}
